// FuseMoEFusion_80092550135869
// MI455X (gfx1250) — matrix operand order repaired and checked by reading, not compiled, not run on hardware
//
#include <hip/hip_runtime.h>
#include <math.h>

// ---------------------------------------------------------------------------
// MI455X (gfx1250) fused MoE: bf16 WMMA expert GEMMs (4-way N-blocked),
// top-2 dispatch, async LDS gathers (GLOBAL_LOAD_ASYNC_TO_LDS_B128),
// weights live in L2 (128MB bf16 < 192MB), LDS-resident hidden activations.
// ---------------------------------------------------------------------------

typedef __attribute__((ext_vector_type(16))) __bf16 v16bf;
typedef __attribute__((ext_vector_type(8)))  __bf16 v8bf;
typedef __attribute__((ext_vector_type(8)))  float  v8f;

#define BTOK   2048
#define NMOD   4
#define DMODEL 1024
#define NEXP   8
#define HDIM   4096
#define TROWS  (BTOK * NMOD)        // 8192 token rows
#define GATEIN (NMOD * DMODEL)      // 4096
#define MAXTILES 1032               // 16384/16 + NEXP padding tiles
#define ROWCAP   (MAXTILES * 16)

// ---- bf16 helpers (manual RNE convert; no reliance on native __bf16 arith) --
__device__ inline __bf16 f2bf(float f) {
  unsigned u = __builtin_bit_cast(unsigned, f);
  unsigned r = (u + 0x7FFFu + ((u >> 16) & 1u)) >> 16;
  unsigned short s = (unsigned short)r;
  return __builtin_bit_cast(__bf16, s);
}
__device__ inline float bf2f(__bf16 b) {
  unsigned u = ((unsigned)__builtin_bit_cast(unsigned short, b)) << 16;
  return __builtin_bit_cast(float, u);
}

// ---- CDNA5 async copy: global -> LDS, 16B per lane, ASYNCcnt-tracked -------
__device__ inline void async_copy16(unsigned ldsOff, unsigned long long gaddr) {
  asm volatile("global_load_async_to_lds_b128 %0, %1, off"
               :: "v"(ldsOff), "v"(gaddr) : "memory");
}
__device__ inline void async_wait0() {
  asm volatile("s_wait_asynccnt 0x0" ::: "memory");
}

// A fragment (16x32 bf16 tile, row-major in LDS, leading dim ld):
// lane l<16 : row l,    K = k..k+15
// lane l>=16: row l-16, K = k+16..k+31   (kb folds the +16)
__device__ inline v16bf load_a_frag(const __bf16* tile, int row, int k, int ld) {
  const __bf16* p = tile + row * ld + k;
  v8bf c0 = *reinterpret_cast<const v8bf*>(p);
  v8bf c1 = *reinterpret_cast<const v8bf*>(p + 8);
  v16bf r;
#pragma unroll
  for (int i = 0; i < 8; ++i) { r[i] = c0[i]; r[i + 8] = c1[i]; }
  return r;
}
// B fragment (32x16): lane holds column n=l&15, 16 contiguous K from WT row.
__device__ inline v16bf load_b_frag(const __bf16* p) {
  return *reinterpret_cast<const v16bf*>(p);
}
// reuse_a is an OPSEL immediate -> must be a compile-time constant.
template <bool RA>
__device__ inline v8f wmma_bf16(v16bf a, v16bf b, v8f c) {
  return __builtin_amdgcn_wmma_f32_16x16x32_bf16(false, a, false, b,
                                                 (short)0, c, RA, false);
}

// ---------------------------- utility kernels ------------------------------
__global__ void fill_f32_kernel(float* p, float v, int n) {
  for (int i = blockIdx.x * blockDim.x + threadIdx.x; i < n;
       i += gridDim.x * blockDim.x) p[i] = v;
}
__global__ void fill_i32_kernel(int* p, int v, int n) {
  for (int i = blockIdx.x * blockDim.x + threadIdx.x; i < n;
       i += gridDim.x * blockDim.x) p[i] = v;
}
__global__ void cast_bf16_kernel(const float* __restrict__ src,
                                 __bf16* __restrict__ dst, int n) {
  for (int i = blockIdx.x * blockDim.x + threadIdx.x; i < n;
       i += gridDim.x * blockDim.x) dst[i] = f2bf(src[i]);
}
// dst[c*R + r] = bf16(src[r*C + c]); dims multiples of 32. blockDim=(32,8).
__global__ void transpose_cast_kernel(const float* __restrict__ src,
                                      __bf16* __restrict__ dst, int R, int C) {
  __shared__ float tile[32][33];
  size_t batch = (size_t)blockIdx.z * R * C;
  src += batch; dst += batch;
  int c0 = blockIdx.x * 32, r0 = blockIdx.y * 32;
  int tx = threadIdx.x, ty = threadIdx.y;
#pragma unroll
  for (int i = 0; i < 32; i += 8)
    tile[ty + i][tx] = src[(size_t)(r0 + ty + i) * C + (c0 + tx)];
  __syncthreads();
#pragma unroll
  for (int i = 0; i < 32; i += 8)
    dst[(size_t)(c0 + ty + i) * R + (r0 + tx)] = f2bf(tile[tx][ty + i]);
}

// ------------------------- projection GEMM (WMMA) --------------------------
// tok[(b*NMOD+mod)*D + n] = x[b,:] @ Wp[:,n] + bp[n]; 16 tokens per block.
__global__ __launch_bounds__(256) void proj_wmma_kernel(
    const __bf16* __restrict__ xbf, const __bf16* __restrict__ WpT,
    const float* __restrict__ bp, float* __restrict__ tok, int dm, int mod) {
  extern __shared__ char smem[];
  __bf16* At = (__bf16*)smem;                 // 16 x dm bf16
  int tid = threadIdx.x;
  int b0 = blockIdx.x * 16;
  {   // async-gather 16 rows of x (bf16) into LDS
    int r = tid >> 4, seg = tid & 15;
    int per = dm >> 7;                        // 16B chunks per segment
    unsigned long long g =
        (unsigned long long)(const char*)(xbf + (size_t)(b0 + r) * dm);
    unsigned l = (unsigned)(unsigned long long)(char*)(At + r * dm);
    for (int i = 0; i < per; ++i) {
      unsigned byteOff = (unsigned)(seg * per + i) * 16u;
      async_copy16(l + byteOff, g + byteOff);
    }
  }
  async_wait0();
  __syncthreads();
  int wave = tid >> 5, lane = tid & 31;
  int row = lane & 15, hi = lane >> 4, kb = hi * 16, nlo = lane & 15;
  for (int g4 = 0; g4 < 2; ++g4) {            // 64 N-tiles / 8 waves / 4-block
    int nt0 = wave * 8 + g4 * 4;
    v8f acc[4];
#pragma unroll
    for (int j = 0; j < 4; ++j)
      acc[j] = (v8f){0.f,0.f,0.f,0.f,0.f,0.f,0.f,0.f};
    int n0 = nt0 * 16 + nlo;
    const __bf16* B0 = WpT + (size_t)n0 * dm + hi * 16;
    for (int k0 = 0; k0 < dm; k0 += 32) {
      v16bf a = load_a_frag(At, row, kb + k0, dm);
      acc[0] = wmma_bf16<false>(a, load_b_frag(B0 + k0),                  acc[0]);
      acc[1] = wmma_bf16<true >(a, load_b_frag(B0 + (size_t)16 * dm + k0), acc[1]);
      acc[2] = wmma_bf16<true >(a, load_b_frag(B0 + (size_t)32 * dm + k0), acc[2]);
      acc[3] = wmma_bf16<true >(a, load_b_frag(B0 + (size_t)48 * dm + k0), acc[3]);
    }
#pragma unroll
    for (int j = 0; j < 4; ++j) {
      int ncol = n0 + j * 16;
      float bias = bp[ncol];
#pragma unroll
      for (int i = 0; i < 8; ++i) {
        int m = i + 8 * hi;                    // token within tile
        size_t t = (size_t)(b0 + m) * NMOD + mod;
        tok[t * DMODEL + ncol] = acc[j][i] + bias;
      }
    }
  }
}

// ------------------------------ LayerNorm ----------------------------------
__global__ void ln_kernel(float* __restrict__ tok, __bf16* __restrict__ tokbf,
                          const float* __restrict__ g, const float* __restrict__ b) {
  __shared__ float rs[256], rs2[256];
  int row = blockIdx.x, tid = threadIdx.x;
  float* p = tok + (size_t)row * DMODEL;
  float s = 0.f, s2 = 0.f;
  for (int j = tid; j < DMODEL; j += 256) { float v = p[j]; s += v; s2 += v * v; }
  rs[tid] = s; rs2[tid] = s2; __syncthreads();
  for (int o = 128; o > 0; o >>= 1) {
    if (tid < o) { rs[tid] += rs[tid + o]; rs2[tid] += rs2[tid + o]; }
    __syncthreads();
  }
  float mean = rs[0] * (1.f / DMODEL);
  float var = rs2[0] * (1.f / DMODEL) - mean * mean;
  float rstd = rsqrtf(var + 1e-5f);
  for (int j = tid; j < DMODEL; j += 256) {
    float v = (p[j] - mean) * rstd * g[j] + b[j];
    p[j] = v;
    tokbf[(size_t)row * DMODEL + j] = f2bf(v);
  }
}

// --------------------------- router / top-2 gate ---------------------------
__global__ void gate_kernel(const float* __restrict__ ctx, const float* __restrict__ Wg,
                            const float* __restrict__ bg, float* __restrict__ gates,
                            int* __restrict__ eidx, float* __restrict__ ew) {
  __shared__ float red[256 * NEXP];
  int b = blockIdx.x, tid = threadIdx.x;
  float p[NEXP];
#pragma unroll
  for (int e = 0; e < NEXP; ++e) p[e] = 0.f;
  const float* c = ctx + (size_t)b * GATEIN;
  for (int j = tid; j < GATEIN; j += 256) {
    float cv = c[j];
    const float* wr = Wg + (size_t)j * NEXP;
#pragma unroll
    for (int e = 0; e < NEXP; ++e) p[e] += cv * wr[e];
  }
#pragma unroll
  for (int e = 0; e < NEXP; ++e) red[tid * NEXP + e] = p[e];
  __syncthreads();
  for (int o = 128; o > 0; o >>= 1) {
    if (tid < o)
#pragma unroll
      for (int e = 0; e < NEXP; ++e) red[tid * NEXP + e] += red[(tid + o) * NEXP + e];
    __syncthreads();
  }
  if (tid == 0) {
    float lg[NEXP];
#pragma unroll
    for (int e = 0; e < NEXP; ++e) lg[e] = red[e] + bg[e];
    int i0 = 0;
    for (int e = 1; e < NEXP; ++e) if (lg[e] > lg[i0]) i0 = e;
    int i1 = -1;
    for (int e = 0; e < NEXP; ++e)
      if (e != i0 && (i1 < 0 || lg[e] > lg[i1])) i1 = e;
    float m = fmaxf(lg[i0], lg[i1]);
    float e0 = __expf(lg[i0] - m), e1 = __expf(lg[i1] - m);
    float inv = 1.f / (e0 + e1);
    float w0 = e0 * inv, w1 = e1 * inv;
    for (int e = 0; e < NEXP; ++e)
      gates[(size_t)b * NEXP + e] = (e == i0) ? w0 : ((e == i1) ? w1 : 0.f);
    eidx[b * 2] = i0; eidx[b * 2 + 1] = i1;
    ew[b * 2] = w0;  ew[b * 2 + 1] = w1;
  }
}

// ---------------------------- MoE dispatch ---------------------------------
__global__ void count_kernel(const int* __restrict__ eidx, int* __restrict__ cnt) {
  int b = blockIdx.x * blockDim.x + threadIdx.x;
  if (b < BTOK) {
    atomicAdd(&cnt[eidx[b * 2]], NMOD);
    atomicAdd(&cnt[eidx[b * 2 + 1]], NMOD);
  }
}
__global__ void scan_kernel(const int* __restrict__ cnt, int* __restrict__ tileExpert,
                            int* __restrict__ tileCnt, int* __restrict__ rowBase) {
  if (blockIdx.x == 0 && threadIdx.x == 0) {
    int acc = 0;
    for (int e = 0; e < NEXP; ++e) {
      rowBase[e] = acc * 16;
      int nt = (cnt[e] + 15) >> 4;           // 16-row aligned tiles per expert
      for (int i = 0; i < nt; ++i) tileExpert[acc + i] = e;
      acc += nt;
    }
    tileCnt[0] = acc;
  }
}
__global__ void scatter_kernel(const int* __restrict__ eidx, const float* __restrict__ ew,
                               const int* __restrict__ rowBase, int* __restrict__ cursors,
                               int* __restrict__ rowList, float* __restrict__ rowW) {
  int b = blockIdx.x * blockDim.x + threadIdx.x;
  if (b < BTOK) {
    for (int s = 0; s < 2; ++s) {
      int e = eidx[b * 2 + s];
      int pos = atomicAdd(&cursors[e], NMOD);
      int base = rowBase[e] + pos;
      float w = ew[b * 2 + s];
      for (int mm = 0; mm < NMOD; ++mm) {
        rowList[base + mm] = b * NMOD + mm;
        rowW[base + mm] = w;
      }
    }
  }
}

// -------------------- fused expert FFN (WMMA, LDS-resident h) --------------
// One block = 16 gathered token rows x one expert.
// LDS: A(16x1024 bf16)=32KB, H(16x4096 bf16)=128KB  -> needs CDNA5 320KB/WGP.
__global__ __launch_bounds__(256) void expert_kernel(
    const __bf16* __restrict__ tokbf, const __bf16* __restrict__ W1T,
    const float* __restrict__ b1, const __bf16* __restrict__ W2T,
    const float* __restrict__ b2, const int* __restrict__ tileExpert,
    const int* __restrict__ tileCnt, const int* __restrict__ rowList,
    const float* __restrict__ rowW, float* __restrict__ combined) {
  int tile = blockIdx.x;
  if (tile >= tileCnt[0]) return;
  extern __shared__ char smem[];
  __bf16* At = (__bf16*)smem;                         // 16 x DMODEL
  __bf16* Ht = (__bf16*)(smem + 32768);               // 16 x HDIM
  int*   rIdx = (int*)(smem + 32768 + 131072);        // 16
  float* rWt  = (float*)(smem + 32768 + 131072 + 64); // 16
  int tid = threadIdx.x;
  int e = tileExpert[tile];
  if (tid < 16) {
    rIdx[tid] = rowList[tile * 16 + tid];
    rWt[tid]  = rowW[tile * 16 + tid];
  }
  __syncthreads();
  {   // async-gather 16 token rows (bf16) into LDS; padded rows (-1) -> zeros
    int r = tid >> 4, seg = tid & 15;
    int t = rIdx[r];
    unsigned l = (unsigned)(unsigned long long)(char*)(At + r * DMODEL) +
                 (unsigned)seg * 128u;
    if (t >= 0) {
      unsigned long long g =
          (unsigned long long)(const char*)(tokbf + (size_t)t * DMODEL) +
          (unsigned long long)seg * 128ull;
#pragma unroll
      for (int i = 0; i < 8; ++i)
        async_copy16(l + (unsigned)i * 16u, g + (unsigned long long)i * 16ull);
    } else {
      uint4* d = reinterpret_cast<uint4*>(At + r * DMODEL) + seg * 8;
      uint4 z = make_uint4(0u, 0u, 0u, 0u);
#pragma unroll
      for (int i = 0; i < 8; ++i) d[i] = z;
    }
  }
  async_wait0();
  __syncthreads();
  int wave = tid >> 5, lane = tid & 31;
  int row = lane & 15, hi = lane >> 4, kb = hi * 16, nlo = lane & 15;

  // GEMM1: h = gelu(tokens @ W1[e] + b1[e]); 256 N-tiles over H, K=1024.
  // 4-way N-blocking: one A-fragment LDS load feeds 4 WMMAs (reuse_a hint).
  const __bf16* W1e = W1T + (size_t)e * HDIM * DMODEL;
  const float*  b1e = b1 + (size_t)e * HDIM;
  for (int g4 = 0; g4 < 8; ++g4) {
    int nt0 = wave * 32 + g4 * 4;
    v8f acc[4];
#pragma unroll
    for (int j = 0; j < 4; ++j)
      acc[j] = (v8f){0.f,0.f,0.f,0.f,0.f,0.f,0.f,0.f};
    int h0 = nt0 * 16 + nlo;
    const __bf16* B0 = W1e + (size_t)h0 * DMODEL + hi * 16;
    for (int k0 = 0; k0 < DMODEL; k0 += 32) {
      v16bf a = load_a_frag(At, row, kb + k0, DMODEL);
      acc[0] = wmma_bf16<false>(a, load_b_frag(B0 + k0),               acc[0]);
      acc[1] = wmma_bf16<true >(a, load_b_frag(B0 + 16 * DMODEL + k0), acc[1]);
      acc[2] = wmma_bf16<true >(a, load_b_frag(B0 + 32 * DMODEL + k0), acc[2]);
      acc[3] = wmma_bf16<true >(a, load_b_frag(B0 + 48 * DMODEL + k0), acc[3]);
    }
#pragma unroll
    for (int j = 0; j < 4; ++j) {
      int h = h0 + j * 16;
      float bias = b1e[h];
#pragma unroll
      for (int i = 0; i < 8; ++i) {
        float v = acc[j][i] + bias;
        v = 0.5f * v * (1.0f + erff(v * 0.70710678118654752f));   // exact gelu
        Ht[(i + 8 * hi) * HDIM + h] = f2bf(v);
      }
    }
  }
  __syncthreads();

  // GEMM2: out = h @ W2[e] + b2[e]; 64 N-tiles over D, K=4096. Same blocking.
  const __bf16* W2e = W2T + (size_t)e * DMODEL * HDIM;
  const float*  b2e = b2 + (size_t)e * DMODEL;
  for (int g4 = 0; g4 < 2; ++g4) {
    int nt0 = wave * 8 + g4 * 4;
    v8f acc[4];
#pragma unroll
    for (int j = 0; j < 4; ++j)
      acc[j] = (v8f){0.f,0.f,0.f,0.f,0.f,0.f,0.f,0.f};
    int d0 = nt0 * 16 + nlo;
    const __bf16* B0 = W2e + (size_t)d0 * HDIM + hi * 16;
    for (int k0 = 0; k0 < HDIM; k0 += 32) {
      v16bf a = load_a_frag(Ht, row, kb + k0, HDIM);
      acc[0] = wmma_bf16<false>(a, load_b_frag(B0 + k0),             acc[0]);
      acc[1] = wmma_bf16<true >(a, load_b_frag(B0 + 16 * HDIM + k0), acc[1]);
      acc[2] = wmma_bf16<true >(a, load_b_frag(B0 + 32 * HDIM + k0), acc[2]);
      acc[3] = wmma_bf16<true >(a, load_b_frag(B0 + 48 * HDIM + k0), acc[3]);
    }
#pragma unroll
    for (int j = 0; j < 4; ++j) {
      int dcol = d0 + j * 16;
      float bias = b2e[dcol];
#pragma unroll
      for (int i = 0; i < 8; ++i) {
        int m = i + 8 * hi;
        int t = rIdx[m];
        if (t >= 0)
          atomicAdd(&combined[(size_t)t * DMODEL + dcol],
                    (acc[j][i] + bias) * rWt[m]);
      }
    }
  }
}

// ------------------------------ pool + loss --------------------------------
__global__ void pooled_kernel(const float* __restrict__ combined,
                              float* __restrict__ pooled) {
  int b = blockIdx.x, tid = threadIdx.x;
  for (int j = tid; j < DMODEL; j += 256) {
    float s = 0.f;
#pragma unroll
    for (int m = 0; m < NMOD; ++m)
      s += combined[(size_t)(b * NMOD + m) * DMODEL + j];
    pooled[(size_t)b * DMODEL + j] = s * 0.25f;
  }
}
__global__ void balance_kernel(const float* __restrict__ gates,
                               float* __restrict__ out) {
  __shared__ float r1[NEXP * 256], r2[NEXP * 256];
  int tid = threadIdx.x;
  float imp[NEXP], ld[NEXP];
#pragma unroll
  for (int e = 0; e < NEXP; ++e) { imp[e] = 0.f; ld[e] = 0.f; }
  for (int b = tid; b < BTOK; b += 256)
#pragma unroll
    for (int e = 0; e < NEXP; ++e) {
      float g = gates[(size_t)b * NEXP + e];
      imp[e] += g;
      ld[e]  += (g > 0.f) ? 1.f : 0.f;
    }
#pragma unroll
  for (int e = 0; e < NEXP; ++e) { r1[e * 256 + tid] = imp[e]; r2[e * 256 + tid] = ld[e]; }
  __syncthreads();
  for (int o = 128; o > 0; o >>= 1) {
    if (tid < o)
#pragma unroll
      for (int e = 0; e < NEXP; ++e) {
        r1[e * 256 + tid] += r1[e * 256 + tid + o];
        r2[e * 256 + tid] += r2[e * 256 + tid + o];
      }
    __syncthreads();
  }
  if (tid == 0) {
    float mi = 0.f, ml = 0.f;
    for (int e = 0; e < NEXP; ++e) { mi += r1[e * 256]; ml += r2[e * 256]; }
    mi *= (1.f / NEXP); ml *= (1.f / NEXP);
    float vi = 0.f, vl = 0.f;
    for (int e = 0; e < NEXP; ++e) {
      float di = r1[e * 256] - mi, dl = r2[e * 256] - ml;
      vi += di * di; vl += dl * dl;
    }
    vi *= (1.f / (NEXP - 1)); vl *= (1.f / (NEXP - 1));   // ddof=1
    float cvi = vi / (mi * mi + 1e-10f);
    float cvl = vl / (ml * ml + 1e-10f);
    out[0] = 0.01f * (cvi + cvl);
  }
}

// ------------------------------- launcher ----------------------------------
extern "C" void kernel_launch(void* const* d_in, const int* in_sizes, int n_in,
                              void* d_out, int out_size, void* d_ws, size_t ws_size,
                              hipStream_t stream) {
  (void)in_sizes; (void)n_in; (void)out_size; (void)ws_size;
  static const int DIMS[NMOD] = {768, 1536, 512, 2048};
  // inputs in setup_inputs() dict order
  const float* x[NMOD]  = {(const float*)d_in[0], (const float*)d_in[3],
                           (const float*)d_in[6], (const float*)d_in[9]};
  const float* Wp[NMOD] = {(const float*)d_in[1], (const float*)d_in[4],
                           (const float*)d_in[7], (const float*)d_in[10]};
  const float* bp[NMOD] = {(const float*)d_in[2], (const float*)d_in[5],
                           (const float*)d_in[8], (const float*)d_in[11]};
  const float* ln_g = (const float*)d_in[12];
  const float* ln_b = (const float*)d_in[13];
  const float* Wg = (const float*)d_in[14];
  const float* bg = (const float*)d_in[15];
  const float* W1 = (const float*)d_in[16];
  const float* b1 = (const float*)d_in[17];
  const float* W2 = (const float*)d_in[18];
  const float* b2 = (const float*)d_in[19];

  float* out_pooled   = (float*)d_out;                        // [B, D]
  float* out_combined = out_pooled + (size_t)BTOK * DMODEL;   // [B, M, D]
  float* out_gates    = out_combined + (size_t)TROWS * DMODEL;// [B, E]
  float* out_bal      = out_gates + (size_t)BTOK * NEXP;      // [1]

  // workspace carve-up (256B aligned)
  char* base = (char*)d_ws;
  size_t off = 0;
  auto wsa = [&](size_t bytes) -> char* {
    char* p = base + off;
    off += (bytes + 255) & ~(size_t)255;
    return p;
  };
  float*  tok_f32 = (float*)wsa((size_t)TROWS * DMODEL * 4);
  __bf16* tok_bf  = (__bf16*)wsa((size_t)TROWS * DMODEL * 2);
  int dimsum = 0; for (int m = 0; m < NMOD; ++m) dimsum += DIMS[m];
  __bf16* xbf  = (__bf16*)wsa((size_t)BTOK * dimsum * 2);
  __bf16* WpT  = (__bf16*)wsa((size_t)DMODEL * dimsum * 2);
  __bf16* W1T  = (__bf16*)wsa((size_t)NEXP * HDIM * DMODEL * 2);
  __bf16* W2T  = (__bf16*)wsa((size_t)NEXP * DMODEL * HDIM * 2);
  int*    eidx = (int*)wsa((size_t)BTOK * 2 * 4);
  float*  ew   = (float*)wsa((size_t)BTOK * 2 * 4);
  int*    cnt     = (int*)wsa(NEXP * 4);
  int*    cursors = (int*)wsa(NEXP * 4);
  int*    rowBase = (int*)wsa(NEXP * 4);
  int*    tileExp = (int*)wsa(MAXTILES * 4);
  int*    tileCnt = (int*)wsa(4);
  int*    rowList = (int*)wsa((size_t)ROWCAP * 4);
  float*  rowW    = (float*)wsa((size_t)ROWCAP * 4);

  // 0) init state that must be zero / invalid each call
  fill_f32_kernel<<<2048, 256, 0, stream>>>(out_combined, 0.f, TROWS * DMODEL);
  fill_i32_kernel<<<1, 64, 0, stream>>>(cnt, 0, NEXP);
  fill_i32_kernel<<<1, 64, 0, stream>>>(cursors, 0, NEXP);
  fill_i32_kernel<<<65, 256, 0, stream>>>(rowList, -1, ROWCAP);

  // 1) precision/layout conversion: x -> bf16; Wp/W1/W2 -> transposed bf16
  {
    int xo = 0, wo = 0;
    for (int m = 0; m < NMOD; ++m) {
      int nx = BTOK * DIMS[m];
      cast_bf16_kernel<<<(nx + 255) / 256, 256, 0, stream>>>(x[m], xbf + (size_t)xo, nx);
      transpose_cast_kernel<<<dim3(DMODEL / 32, DIMS[m] / 32, 1), dim3(32, 8, 1),
                              0, stream>>>(Wp[m], WpT + (size_t)wo, DIMS[m], DMODEL);
      xo += BTOK * DIMS[m];
      wo += DMODEL * DIMS[m];
    }
  }
  transpose_cast_kernel<<<dim3(HDIM / 32, DMODEL / 32, NEXP), dim3(32, 8, 1),
                          0, stream>>>(W1, W1T, DMODEL, HDIM);   // -> [e][h][d]
  transpose_cast_kernel<<<dim3(DMODEL / 32, HDIM / 32, NEXP), dim3(32, 8, 1),
                          0, stream>>>(W2, W2T, HDIM, DMODEL);   // -> [e][d][h]

  // 2) modality projections (WMMA GEMM, 16-token tiles)
  {
    int xo = 0, wo = 0;
    for (int m = 0; m < NMOD; ++m) {
      proj_wmma_kernel<<<BTOK / 16, 256, (size_t)DIMS[m] * 32, stream>>>(
          xbf + (size_t)xo, WpT + (size_t)wo, bp[m], tok_f32, DIMS[m], m);
      xo += BTOK * DIMS[m];
      wo += DMODEL * DIMS[m];
    }
  }

  // 3) LayerNorm (f32 context for router + bf16 tokens for experts)
  ln_kernel<<<TROWS, 256, 0, stream>>>(tok_f32, tok_bf, ln_g, ln_b);

  // 4) router: logits, top-2 softmax gates
  gate_kernel<<<BTOK, 256, 0, stream>>>(tok_f32, Wg, bg, out_gates, eidx, ew);

  // 5) MoE dispatch: count -> scan -> scatter (16-row aligned expert tiles)
  count_kernel<<<(BTOK + 255) / 256, 256, 0, stream>>>(eidx, cnt);
  scan_kernel<<<1, 1, 0, stream>>>(cnt, tileExp, tileCnt, rowBase);
  scatter_kernel<<<(BTOK + 255) / 256, 256, 0, stream>>>(eidx, ew, rowBase,
                                                         cursors, rowList, rowW);

  // 6) fused expert FFN; worst-case grid (device-side early-out), 160KB LDS
  expert_kernel<<<MAXTILES, 256, 163968, stream>>>(
      tok_bf, W1T, b1, W2T, b2, tileExp, tileCnt, rowList, rowW, out_combined);

  // 7) modality-mean pool + load-balance loss
  pooled_kernel<<<BTOK, 256, 0, stream>>>(out_combined, out_pooled);
  balance_kernel<<<1, 256, 0, stream>>>(out_gates, out_bal);
}
